// DConv_19859928776736
// MI455X (gfx1250) — matrix operand order repaired and checked by reading, not compiled, not run on hardware
//
#include <hip/hip_runtime.h>

typedef __attribute__((ext_vector_type(16))) __bf16 v16bf;
typedef __attribute__((ext_vector_type(8)))  float  v8f;

#define BATCH 16
#define CIN   64
#define COUT  64
#define HW    160
#define TR    8     // output rows per block tile
#define TC    16    // output cols per block tile
#define II    10    // staged rows  (TR + 2)
#define JJ    18    // staged cols  (TC + 2)
// per-(ii,jj) LDS block: 64 ci * bf16 hi-plane (128B) + 64 ci * bf16 lo-plane (128B) + 16B pad
#define BLK_H 136   // block stride in halfwords (= 68 dwords = 272B; 4-dword pad -> bank spread)

// ---- fp32 -> bf16 (RNE) and hi/lo split -------------------------------------
__device__ __forceinline__ unsigned f2bf(float f) {
  unsigned u = __float_as_uint(f);
  unsigned r = u + 0x7FFFu + ((u >> 16) & 1u);
  return r >> 16;
}
__device__ __forceinline__ void split_bf(float f, unsigned& h, unsigned& l) {
  h = f2bf(f);
  float hf = __uint_as_float(h << 16);
  l = f2bf(f - hf);
}

// ---- weight pre-pass: fp32 OIHW -> bf16 hi/lo in WMMA A-operand layout ------
// chunk c = (kh*3+kw)*2 + cihalf, c in [0,18)
// A 16x32 bf16 layout (ISA 7.12.2): lane = M (co within 16-row tile),
//   dword v: K pair (K0, K0+1) with K0 = 2v + (v>=4 ? 8 : 0) + (lane>=16 ? 8 : 0),
//   K0 in [15:0], K0+1 in [31:16].
// ws dwords: base = ((c*4 + wm)*32 + lane)*16 ;  [base+v]=hi pair, [base+8+v]=lo pair
__global__ void wprep_kernel(const float* __restrict__ w, unsigned* __restrict__ ws) {
  int t    = blockIdx.x * 256 + threadIdx.x;      // 18*4*32*8 = 18432 total
  int v    = t & 7;
  int lane = (t >> 3) & 31;
  int wm   = (t >> 8) & 3;
  int c    = t >> 10;                             // 0..17
  if (c >= 18) return;
  int ch = c & 1;
  int rk = c >> 1;                                // 0..8 = kh*3+kw
  int kh = rk / 3, kw = rk % 3;
  int K0 = 2 * v + ((v >= 4) ? 8 : 0) + ((lane >= 16) ? 8 : 0);
  int ci = ch * 32 + K0;
  int co = wm * 16 + (lane & 15);
  float w0 = w[((co * CIN + ci    ) * 3 + kh) * 3 + kw];
  float w1 = w[((co * CIN + ci + 1) * 3 + kh) * 3 + kw];
  unsigned h0, l0, h1, l1;
  split_bf(w0, h0, l0);
  split_bf(w1, h1, l1);
  int base = ((c * 4 + wm) * 32 + lane) * 16;
  ws[base + v]     = h0 | (h1 << 16);   // K0 in [15:0], K0+1 in [31:16]
  ws[base + 8 + v] = l0 | (l1 << 16);
}

// ---- main implicit-GEMM conv kernel -----------------------------------------
__global__ __launch_bounds__(256) void dconv_kernel(const float* __restrict__ x,
                                                    const unsigned* __restrict__ ws,
                                                    float* __restrict__ out) {
  // 10*18*272B = 48,960 B of LDS: separate hi/lo bf16 planes, ci innermost
  __shared__ unsigned short smem16[II * JJ * BLK_H];

  const int tid = threadIdx.x;
  const int bid = blockIdx.x;
  const int b   = bid / 200;
  const int rem = bid % 200;
  const int oh0 = (rem / 10) * TR;
  const int ow0 = (rem % 10) * TC;

  // Stage shifted, zero-filled input tile.
  // tile value = X(b, ci, oh0+ii-dy[ci]-1, ow0+jj-dx[ci]-1), zero out of range.
  for (int e = tid; e < CIN * II * JJ; e += 256) {   // 11520 = 45 * 256 exactly
    int ci = e / (II * JJ);
    int r  = e - ci * (II * JJ);
    int ii = r / JJ;
    int jj = r - ii * JJ;
    int m5 = ci % 5;
    int dy = (m5 == 2) ? 1 : ((m5 == 4) ? -1 : 0);
    int dx = (m5 == 1) ? 1 : ((m5 == 3) ? -1 : 0);
    int gr = oh0 + ii - dy - 1;
    int gc = ow0 + jj - dx - 1;
    float val = 0.f;
    if ((unsigned)gr < (unsigned)HW && (unsigned)gc < (unsigned)HW)
      val = x[((b * CIN + ci) * HW + gr) * HW + gc];
    unsigned h, l;
    split_bf(val, h, l);
    int base16 = (ii * JJ + jj) * BLK_H;
    smem16[base16 + ci]      = (unsigned short)h;   // hi plane
    smem16[base16 + 64 + ci] = (unsigned short)l;   // lo plane
  }
  __syncthreads();

  const int lane = tid & 31;
  const int wave = tid >> 5;
  const int wm   = wave & 3;        // M-tile: co = wm*16 + ...
  const int wn   = wave >> 2;       // N-tile: rows wn*4 .. wn*4+3
  const int nn   = lane & 15;       // N within 16 (output column)
  const int hsel = lane >> 4;       // upper half-lanes (K offset +8)

  v8f zero;
#pragma unroll
  for (int g = 0; g < 8; ++g) zero[g] = 0.f;
  v8f acc[4];
#pragma unroll
  for (int s = 0; s < 4; ++s) acc[s] = zero;

  const v16bf* Aop = (const v16bf*)ws;

#pragma unroll 1
  for (int kh = 0; kh < 3; ++kh) {
#pragma unroll 1
    for (int ch = 0; ch < 2; ++ch) {
      // A operands (hi/lo) for the three kw chunks of this (kh, cihalf)
      v16bf ahi[3], alo[3];
#pragma unroll
      for (int kw = 0; kw < 3; ++kw) {
        int c   = (kh * 3 + kw) * 2 + ch;
        int idx = ((c * 4 + wm) * 32 + lane) * 2;   // in 32B units
        ahi[kw] = Aop[idx];
        alo[kw] = Aop[idx + 1];
      }
      const int cib = ch * 32 + hsel * 8;           // halfword (ci) base for this lane's K half
#pragma unroll
      for (int s = 0; s < 4; ++s) {
        const int ii = wn * 4 + s + kh;
#pragma unroll
        for (int kw = 0; kw < 3; ++kw) {
          // B operand: lane = N (column nn at jj = nn+kw); ci cib+0..7 and cib+16..23 bf16,
          // K-pairs already correctly packed (even ci in [15:0]) -> no perms needed.
          const unsigned short* pB = &smem16[(ii * JJ + (nn + kw)) * BLK_H + cib];
          union { uint4 q[2]; v16bf v; } bh, bl;
          bh.q[0] = *(const uint4*)(pB);            // hi plane, 16B aligned
          bh.q[1] = *(const uint4*)(pB + 16);
          bl.q[0] = *(const uint4*)(pB + 64);       // lo plane
          bl.q[1] = *(const uint4*)(pB + 80);
          // fp32-via-bf16x3: hh + hl + lh, accumulated in f32
          acc[s] = __builtin_amdgcn_wmma_f32_16x16x32_bf16(false, ahi[kw], false, bh.v,
                                                           (short)0, acc[s], false, false);
          acc[s] = __builtin_amdgcn_wmma_f32_16x16x32_bf16(false, ahi[kw], false, bl.v,
                                                           (short)0, acc[s], false, false);
          acc[s] = __builtin_amdgcn_wmma_f32_16x16x32_bf16(false, alo[kw], false, bh.v,
                                                           (short)0, acc[s], false, false);
        }
      }
    }
  }

  // Epilogue: C/D layout — VGPR g: lanes 0-15 -> M=g, lanes 16-31 -> M=g+8; lane&15 = N.
  const int co_b = wm * 16 + hsel * 8;
#pragma unroll
  for (int s = 0; s < 4; ++s) {
    const int oh = oh0 + wn * 4 + s;
#pragma unroll
    for (int g = 0; g < 8; ++g) {
      out[((b * COUT + (co_b + g)) * HW + oh) * HW + (ow0 + nn)] = acc[s][g];
    }
  }
}

extern "C" void kernel_launch(void* const* d_in, const int* in_sizes, int n_in,
                              void* d_out, int out_size, void* d_ws, size_t ws_size,
                              hipStream_t stream) {
  const float* x = (const float*)d_in[0];
  const float* w = (const float*)d_in[1];
  float* out     = (float*)d_out;
  unsigned* ws   = (unsigned*)d_ws;   // needs 18*4*32*64 = 147,456 bytes

  // 1) split weights into bf16 hi/lo, pre-arranged in WMMA A-operand layout
  wprep_kernel<<<72, 256, 0, stream>>>(w, ws);
  // 2) implicit-GEMM conv: 16 batches * 20 row-tiles * 10 col-tiles = 3200 blocks
  dconv_kernel<<<BATCH * 20 * 10, 256, 0, stream>>>(x, ws, out);
}
